// RGLRU_73366631350997
// MI455X (gfx1250) — compile-verified, matrix operand order checked by reading, not run on hardware
//
#include <hip/hip_runtime.h>

#define B_DIM 4
#define L_DIM 8192
#define W_DIM 1024
#define H_DIM 8
#define BW    128
#define CHUNK 128
#define NC    (L_DIM / CHUNK)   /* 64 chunks */
#define NTILE (CHUNK / 16)      /* 8 token-tiles per chunk */
#define POWER_F 8.0f

typedef __attribute__((ext_vector_type(16))) __bf16 v16bf;
typedef __attribute__((ext_vector_type(8)))  float  v8f;

static __device__ __forceinline__ unsigned short f2bf_bits(float f) {
  unsigned u = __builtin_bit_cast(unsigned, f);
  unsigned r = u + 0x7FFFu + ((u >> 16) & 1u);   // round-to-nearest-even
  return (unsigned short)(r >> 16);
}
static __device__ __forceinline__ __bf16 bits2bf(unsigned short s) {
  return __builtin_bit_cast(__bf16, s);
}
static __device__ __forceinline__ float sigmoidf_(float v) {
  return 1.0f / (1.0f + __expf(-v));
}

// phase 0: write per-chunk aggregates (A=prod a, S=local scan end) into d_out
//          token slots 1 and 2 of each chunk.
// phase 1: read per-chunk carry from token slot 0, recompute gates, run the
//          seeded local scan and write the final output for every token.
__global__ void __launch_bounds__(256)
rglru_gates_scan(const float* __restrict__ x,
                 const float* __restrict__ a_param,
                 const float* __restrict__ w_in, const float* __restrict__ b_in,
                 const float* __restrict__ w_a,  const float* __restrict__ b_a,
                 float* __restrict__ out, int phase)
{
  const int c  = blockIdx.x;        // chunk
  const int h  = blockIdx.y;        // head
  const int b  = blockIdx.z;        // batch
  const int tid  = threadIdx.x;
  const int wv   = tid >> 5;        // wave id: N-tile = channels [16*wv, 16*wv+16)
  const int ln   = tid & 31;
  const int hi   = (ln >> 4) & 1;   // lane half
  const int nloc = ln & 15;

  __shared__ float          xs [16][BW];  // fp32 x tile (elementwise use)
  __shared__ unsigned short xsb[16][BW];  // bf16 x tile (WMMA A-frags)
  __shared__ float          a_s[16][BW];  // per-token decay
  __shared__ float          z_s[16][BW];  // per-token input / scan output
  __shared__ float          sp_s[BW];     // softplus(a_param) per channel

  if (tid < BW) {
    float ap = a_param[h * BW + tid];
    sp_s[tid] = (ap > 15.0f) ? ap : log1pf(__expf(ap));
  }

  // ---- Weights -> bf16 B-fragments in registers (once per workgroup) ----
  // 16-bit B 32x16 layout (mirror of A 16x32 table, ISA 7.12.2):
  //   lane = N(0..15), elements (2v,2v+1) hold K = k0, k0+1 with
  //   k0 = (v<4 ? 2v : 16+2(v-4)) + 8*laneHalf
  v16bf wfi[4], wfa[4];
  {
    const int col = wv * 16 + nloc;
#pragma unroll
    for (int ks = 0; ks < 4; ++ks) {
#pragma unroll
      for (int v = 0; v < 8; ++v) {
        int k0 = ks * 32 + ((v < 4) ? 2 * v : 16 + 2 * (v - 4)) + hi * 8;
        wfi[ks][2 * v]     = bits2bf(f2bf_bits(w_in[(h * BW + k0)     * BW + col]));
        wfi[ks][2 * v + 1] = bits2bf(f2bf_bits(w_in[(h * BW + k0 + 1) * BW + col]));
        wfa[ks][2 * v]     = bits2bf(f2bf_bits(w_a [(h * BW + k0)     * BW + col]));
        wfa[ks][2 * v + 1] = bits2bf(f2bf_bits(w_a [(h * BW + k0 + 1) * BW + col]));
      }
    }
  }
  const float bi = b_in[h * BW + wv * 16 + nloc];
  const float ba = b_a [h * BW + wv * 16 + nloc];

  float S = 0.0f, Ap = 1.0f, hcar = 0.0f;
  if (phase == 1 && tid < BW)   // carry written by the chunk-scan kernel
    hcar = out[((size_t)b * L_DIM + (size_t)c * CHUNK) * W_DIM + h * BW + tid];

  const long l0 = (long)c * CHUNK;

  for (int t = 0; t < NTILE; ++t) {
    // ---- stage x tile: 16 tokens x 128 channels of head h, coalesced ----
#pragma unroll
    for (int i = 0; i < 8; ++i) {
      int idx = tid + i * 256;
      int tok = idx >> 7, col = idx & 127;
      float xv = x[((size_t)b * L_DIM + l0 + t * 16 + tok) * W_DIM + h * BW + col];
      xs [tok][col] = xv;
      xsb[tok][col] = f2bf_bits(xv);
    }
    __syncthreads();

    // ---- gate matmuls: 2 gates x 4 K-steps of v_wmma_f32_16x16x32_bf16 ----
    v8f acc_i = {}; v8f acc_a = {};
#pragma unroll
    for (int ks = 0; ks < 4; ++ks) {
      v16bf af;                       // A-frag: row M=nloc, K per 16-bit table
      const int m = nloc;
#pragma unroll
      for (int v = 0; v < 8; ++v) {
        int k0 = ks * 32 + ((v < 4) ? 2 * v : 16 + 2 * (v - 4)) + hi * 8;
        af[2 * v]     = bits2bf(xsb[m][k0]);
        af[2 * v + 1] = bits2bf(xsb[m][k0 + 1]);
      }
      acc_i = __builtin_amdgcn_wmma_f32_16x16x32_bf16(false, af, false, wfi[ks],
                                                      (short)0, acc_i, false, false);
      acc_a = __builtin_amdgcn_wmma_f32_16x16x32_bf16(false, af, false, wfa[ks],
                                                      (short)0, acc_a, false, false);
    }

    // ---- elementwise: sigmoid gates, decay a, normalized input z ----
#pragma unroll
    for (int r = 0; r < 8; ++r) {
      int m   = r + 8 * hi;           // f32 C/D layout: VGPR r -> M=r (+8 hi half)
      int chl = wv * 16 + nloc;
      float gx  = sigmoidf_(acc_i[r] + bi);
      float ga  = sigmoidf_(acc_a[r] + ba);
      float la  = -POWER_F * ga * sp_s[chl];
      float av  = __expf(la);
      float mul = sqrtf(fmaxf(0.0f, 1.0f - __expf(2.0f * la)));
      a_s[m][chl] = av;
      z_s[m][chl] = xs[m][chl] * gx * mul;
    }
    __syncthreads();

    // ---- sequential scan across the 16 tokens, one thread per channel ----
    if (tid < BW) {
#pragma unroll
      for (int tt = 0; tt < 16; ++tt) {
        float av = a_s[tt][tid];
        float zv = z_s[tt][tid];
        if (phase == 0) { S = av * S + zv; Ap *= av; }
        else            { hcar = av * hcar + zv; z_s[tt][tid] = hcar; }
      }
    }
    __syncthreads();

    if (phase == 1) {                 // coalesced write-out of the h tile
#pragma unroll
      for (int i = 0; i < 8; ++i) {
        int idx = tid + i * 256;
        int tok = idx >> 7, col = idx & 127;
        out[((size_t)b * L_DIM + l0 + t * 16 + tok) * W_DIM + h * BW + col] =
            z_s[tok][col];
      }
      __syncthreads();
    }
  }

  if (phase == 0 && tid < BW) {       // per-chunk aggregates into slots 1 / 2
    out[((size_t)b * L_DIM + (size_t)c * CHUNK + 1) * W_DIM + h * BW + tid] = Ap;
    out[((size_t)b * L_DIM + (size_t)c * CHUNK + 2) * W_DIM + h * BW + tid] = S;
  }
}

// Pass B: scan the 64 chunk aggregates per (b, channel); deposit each chunk's
// incoming carry into token slot 0 of that chunk.
__global__ void __launch_bounds__(256)
rglru_chunkscan(float* __restrict__ out)
{
  int gid = blockIdx.x * 256 + threadIdx.x;   // 0..4095
  int b = gid >> 10;
  int w = gid & 1023;
  float Hc = 0.0f;
  for (int c = 0; c < NC; ++c) {
    size_t base = ((size_t)b * L_DIM + (size_t)c * CHUNK) * W_DIM + w;
    float Av = out[base + 1 * (size_t)W_DIM];
    float Sv = out[base + 2 * (size_t)W_DIM];
    out[base] = Hc;                   // carry entering chunk c
    Hc = Av * Hc + Sv;
  }
}

extern "C" void kernel_launch(void* const* d_in, const int* in_sizes, int n_in,
                              void* d_out, int out_size, void* d_ws, size_t ws_size,
                              hipStream_t stream) {
  (void)in_sizes; (void)n_in; (void)out_size; (void)d_ws; (void)ws_size;
  const float* x       = (const float*)d_in[0];
  const float* a_param = (const float*)d_in[1];
  const float* w_in    = (const float*)d_in[2];
  const float* b_in    = (const float*)d_in[3];
  const float* w_a     = (const float*)d_in[4];
  const float* b_a     = (const float*)d_in[5];
  float* out = (float*)d_out;

  dim3 grid(NC, H_DIM, B_DIM);        // 64 x 8 x 4 = 2048 workgroups
  rglru_gates_scan<<<grid, 256, 0, stream>>>(x, a_param, w_in, b_in, w_a, b_a, out, 0);
  rglru_chunkscan<<<16, 256, 0, stream>>>(out);
  rglru_gates_scan<<<grid, 256, 0, stream>>>(x, a_param, w_in, b_in, w_a, b_a, out, 1);
}
